// LlamaAttention_48816598286577
// MI455X (gfx1250) — matrix operand order repaired and checked by reading, not compiled, not run on hardware
//
#include <hip/hip_runtime.h>
#include <cstdint>
#include <cstddef>

typedef __attribute__((ext_vector_type(16))) __bf16 v16bf;
typedef __attribute__((ext_vector_type(8)))  float  v8f;

namespace {

constexpr int kS    = 2048;
constexpr int kH    = 4096;
constexpr int kNQ   = 32;
constexpr int kNKV  = 8;
constexpr int kHD   = 128;
constexpr int kQD   = kNQ * kHD;   // 4096
constexpr int kKVD  = kNKV * kHD;  // 1024
constexpr int kBlk  = 128;
constexpr int kSink = 1;
constexpr int kLocal = 8;

struct alignas(16) U128 { unsigned int x, y, z, w; };

__device__ __forceinline__ unsigned short f32_to_bf16(float f) {
  unsigned int u = __builtin_bit_cast(unsigned int, f);
  u += 0x7FFFu + ((u >> 16) & 1u);   // round-to-nearest-even
  return (unsigned short)(u >> 16);
}
__device__ __forceinline__ float bf16_to_f32(unsigned short h) {
  unsigned int u = ((unsigned int)h) << 16;
  return __builtin_bit_cast(float, u);
}

union FragU { U128 h[2]; v16bf v; };

// A fragment: 16 rows x 32 K of bf16 from row-major [row][k] storage.
// Wave32 layout: lanes 0-15 = rows 0-15 holding K {0..7,16..23};
//                lanes 16-31 = rows 0-15 holding K {8..15,24..31}.
__device__ __forceinline__ v16bf frag_a(const unsigned short* base, int stride,
                                        int mbase, int kbase, int lane) {
  const int m    = mbase + (lane & 15);
  const int half = (lane >> 4) & 1;
  const unsigned short* p = base + m * stride + kbase;
  FragU u;
  u.h[0] = *(const U128*)(p + half * 8);
  u.h[1] = *(const U128*)(p + 16 + half * 8);
  return u.v;
}

// B fragment: 32 K x 16 cols of bf16 from K-contiguous (transposed) storage
// Bt[col][k]. Lanes 0-15 = cols 0-15 holding K {0..7,16..23}; lanes 16-31 K {8..15,24..31}.
__device__ __forceinline__ v16bf frag_b(const unsigned short* base, int stride,
                                        int nbase, int kbase, int lane) {
  const int n    = nbase + (lane & 15);
  const int half = (lane >> 4) & 1;
  const unsigned short* p = base + n * stride + kbase;
  FragU u;
  u.h[0] = *(const U128*)(p + half * 8);
  u.h[1] = *(const U128*)(p + 16 + half * 8);
  return u.v;
}

__device__ __forceinline__ v8f wmma_bf16(v16bf a, v16bf b, v8f c) {
  return __builtin_amdgcn_wmma_f32_16x16x32_bf16(false, a, false, b,
                                                 (short)0, c, false, false);
}

__device__ __forceinline__ v8f zero8() {
  v8f z = {0.f, 0.f, 0.f, 0.f, 0.f, 0.f, 0.f, 0.f};
  return z;
}

// CDNA5 async global->LDS copy (16B), SADDR form; tracked by ASYNCcnt.
__device__ __forceinline__ void async_ld_b128(unsigned int lds_byte_addr,
                                              unsigned int gbl_byte_off,
                                              unsigned long long gbl_base) {
  asm volatile("global_load_async_to_lds_b128 %0, %1, %2"
               :: "v"(lds_byte_addr), "v"(gbl_byte_off), "s"(gbl_base)
               : "memory");
}
__device__ __forceinline__ void wait_async0() {
  asm volatile("s_wait_asynccnt 0" ::: "memory");
}

} // namespace

// ---------------------------------------------------------------------------
// GEMM: C[M,N] = A[M,K] * B[K,N].  A is f32 or bf16 (ushort), B is f32
// (converted to bf16 while staging to LDS), C is f32 or bf16.
// 128x128 tile, K-step 32, 256 threads = 8 waves, each wave -> 32x64.
// Double-buffered LDS ping-pong (one barrier per K-step): tile kt+1 is staged
// (bf16-A path via global_load_async_to_lds_b128) while tile kt feeds WMMAs.
// Dynamic LDS layout (u16 units): buf b at b*10240: As 5120, Bts 5120.
// ---------------------------------------------------------------------------
template <bool A_BF16, bool OUT_F32>
__global__ __launch_bounds__(256) void gemm_tile128(
    const void* __restrict__ Aptr, const float* __restrict__ Bptr,
    void* __restrict__ Cptr, int M, int N, int K) {
  extern __shared__ unsigned short sm[];
  const int t = threadIdx.x, lane = t & 31, wid = t >> 5;
  const int m0 = blockIdx.y * 128, n0 = blockIdx.x * 128;
  const int wm = (wid & 3) * 32;   // 4 waves along M
  const int wn = (wid >> 2) * 64;  // 2 waves along N
  const unsigned int ldsbase = __builtin_amdgcn_groupstaticsize();

  v8f acc[2][4];
  #pragma unroll
  for (int i = 0; i < 2; ++i)
    #pragma unroll
    for (int j = 0; j < 4; ++j) acc[i][j] = zero8();

  auto load_tile = [&](int kt, int buf) {
    const int k0 = kt * 32;
    unsigned short* Asb = sm + buf * 10240;
    unsigned short* Btb = sm + buf * 10240 + 5120;
    if constexpr (A_BF16) {
      const unsigned long long ga = (unsigned long long)(uintptr_t)(
          (const unsigned short*)Aptr + (size_t)m0 * K + k0);
      #pragma unroll
      for (int p = 0; p < 2; ++p) {
        const int r  = p * 64 + (t >> 2);
        const int c8 = (t & 3) * 8;
        const unsigned int lds =
            ldsbase + (unsigned)(buf * 10240 + r * 40 + c8) * 2u;
        const unsigned int go =
            ((unsigned)r * (unsigned)K + (unsigned)c8) * 2u;
        async_ld_b128(lds, go, ga);
      }
    } else {
      const float* A = (const float*)Aptr;
      #pragma unroll
      for (int p = 0; p < 4; ++p) {
        const int r  = p * 32 + (t >> 3);
        const int c4 = (t & 7) * 4;
        const float4 f = *(const float4*)&A[(size_t)(m0 + r) * K + k0 + c4];
        Asb[r * 40 + c4 + 0] = f32_to_bf16(f.x);
        Asb[r * 40 + c4 + 1] = f32_to_bf16(f.y);
        Asb[r * 40 + c4 + 2] = f32_to_bf16(f.z);
        Asb[r * 40 + c4 + 3] = f32_to_bf16(f.w);
      }
    }
    #pragma unroll
    for (int p = 0; p < 4; ++p) {
      const int kr = p * 8 + (t >> 5);
      const int c4 = (t & 31) * 4;
      const float4 f = *(const float4*)&Bptr[(size_t)(k0 + kr) * N + n0 + c4];
      Btb[(c4 + 0) * 40 + kr] = f32_to_bf16(f.x);
      Btb[(c4 + 1) * 40 + kr] = f32_to_bf16(f.y);
      Btb[(c4 + 2) * 40 + kr] = f32_to_bf16(f.z);
      Btb[(c4 + 3) * 40 + kr] = f32_to_bf16(f.w);
    }
  };

  const int nkt = K >> 5;
  load_tile(0, 0);
  if constexpr (A_BF16) wait_async0();
  __syncthreads();

  for (int kt = 0; kt < nkt; ++kt) {
    if (kt + 1 < nkt) load_tile(kt + 1, (kt + 1) & 1);
    if (kt + 2 < nkt) {  // global_prefetch_b8 two tiles ahead
      const int k0n = (kt + 2) * 32;
      if constexpr (A_BF16) {
        const unsigned short* A = (const unsigned short*)Aptr;
        __builtin_prefetch(&A[(size_t)(m0 + (t >> 2)) * K + k0n + (t & 3) * 8],
                           0, 0);
      } else {
        const float* A = (const float*)Aptr;
        __builtin_prefetch(&A[(size_t)(m0 + (t >> 3)) * K + k0n + (t & 7) * 4],
                           0, 0);
      }
      __builtin_prefetch(&Bptr[(size_t)(k0n + (t >> 5)) * N + n0 + (t & 31) * 4],
                         0, 0);
    }

    const unsigned short* Asb = sm + (kt & 1) * 10240;
    const unsigned short* Btb = sm + (kt & 1) * 10240 + 5120;
    v16bf af[2], bfr[4];
    #pragma unroll
    for (int i = 0; i < 2; ++i) af[i] = frag_a(Asb, 40, wm + i * 16, 0, lane);
    #pragma unroll
    for (int j = 0; j < 4; ++j) bfr[j] = frag_b(Btb, 40, wn + j * 16, 0, lane);
    #pragma unroll
    for (int i = 0; i < 2; ++i)
      #pragma unroll
      for (int j = 0; j < 4; ++j)
        acc[i][j] = wmma_bf16(af[i], bfr[j], acc[i][j]);

    if constexpr (A_BF16) wait_async0();  // next-tile async writes complete
    __syncthreads();                      // one barrier per K-step
  }

  // C/D layout: elem r of v8f -> row (r + 8*(lane>=16)), col (lane&15)
  const int rh = ((lane >> 4) & 1) * 8;
  const int cn = lane & 15;
  #pragma unroll
  for (int i = 0; i < 2; ++i)
    #pragma unroll
    for (int j = 0; j < 4; ++j)
      #pragma unroll
      for (int r = 0; r < 8; ++r) {
        const int row = m0 + wm + i * 16 + rh + r;
        const int col = n0 + wn + j * 16 + cn;
        if constexpr (OUT_F32)
          ((float*)Cptr)[(size_t)row * N + col] = acc[i][j][r];
        else
          ((unsigned short*)Cptr)[(size_t)row * N + col] =
              f32_to_bf16(acc[i][j][r]);
      }
}

// ---------------------------------------------------------------------------
// RoPE (in place, bf16). One thread per (s, head, i<64) rotation pair.
// ---------------------------------------------------------------------------
__global__ __launch_bounds__(256) void rope_kernel(unsigned short* X,
                                                   int nheads) {
  const int idx  = blockIdx.x * 256 + threadIdx.x;
  const int i    = idx & 63;
  const int rest = idx >> 6;
  const int hh   = rest % nheads;
  const int s    = rest / nheads;
  if (s >= kS) return;
  const float inv = __expf(-(float)i * 0.14391156831212787f);  // ln(1e4)/64
  const float f   = (float)s * inv;
  const float c = __cosf(f), sn = __sinf(f);
  const size_t base = (size_t)s * (nheads * kHD) + (size_t)hh * kHD + i;
  const float x1 = bf16_to_f32(X[base]);
  const float x2 = bf16_to_f32(X[base + 64]);
  X[base]      = f32_to_bf16(x1 * c - x2 * sn);
  X[base + 64] = f32_to_bf16(x2 * c + x1 * sn);
}

// ---------------------------------------------------------------------------
// Flash attention with streaming mask. Grid (qblock=16, head=32), 256 threads.
// Dynamic LDS: Qs/Ks/Vts/Ps, each 128x136 bf16 (139,264 bytes total; CDNA5
// WGP has 320KB LDS). Wave w owns query rows [w*16, w*16+16).
// Q and K tiles are copied with global_load_async_to_lds_b128 (ASYNCcnt);
// V is transposed through VGPRs while staging.
// ---------------------------------------------------------------------------
__global__ __launch_bounds__(256) void attn_kernel(
    const unsigned short* __restrict__ Qb, const unsigned short* __restrict__ Kb,
    const unsigned short* __restrict__ Vb, unsigned short* __restrict__ Ob) {
  extern __shared__ unsigned short smem[];
  constexpr int QS_OFF = 0;              // [q=128][hd=128+pad]
  constexpr int KS_OFF = 128 * 136;      // [key][hd]  (B-frag src for QK^T)
  constexpr int VT_OFF = 2 * 128 * 136;  // [hd][key]  (B-frag src for PV)
  constexpr int PS_OFF = 3 * 128 * 136;  // [q][key]   (A-frag src for PV)
  unsigned short* Qs  = smem + QS_OFF;
  unsigned short* Ks  = smem + KS_OFF;
  unsigned short* Vts = smem + VT_OFF;
  unsigned short* Ps  = smem + PS_OFF;
  const unsigned int ldsbase = __builtin_amdgcn_groupstaticsize();

  const int t = threadIdx.x, lane = t & 31, wid = t >> 5;
  const int qb = blockIdx.x, h = blockIdx.y, kvh = h >> 2;
  const int rh = ((lane >> 4) & 1) * 8;
  const int cn = lane & 15;
  const float scale = 0.088388347648318447f;  // 1/sqrt(128)

  // Q tile: async global->LDS (waited before first use inside the kb loop).
  {
    const unsigned long long gq = (unsigned long long)(uintptr_t)(
        Qb + (size_t)(qb * 128) * kQD + (size_t)h * kHD);
    #pragma unroll
    for (int p = 0; p < 8; ++p) {
      const int r  = p * 16 + (t >> 4);
      const int c8 = (t & 15) * 8;
      async_ld_b128(ldsbase + (unsigned)(QS_OFF + r * 136 + c8) * 2u,
                    ((unsigned)r * (unsigned)kQD + (unsigned)c8) * 2u, gq);
    }
  }

  v8f accO[8];
  float mrow[8], lrow[8];
  for (int i = 0; i < 8; ++i) { accO[i] = zero8(); mrow[i] = -3.0e38f; lrow[i] = 0.f; }

  for (int kb = 0; kb <= qb; ++kb) {
    if (!(kb < kSink || (qb - kb) < kLocal)) continue;  // streaming mask (uniform)
    __syncthreads();  // all waves done reading previous K/V tiles
    {
      const unsigned long long gk = (unsigned long long)(uintptr_t)(
          Kb + (size_t)(kb * 128) * kKVD + (size_t)kvh * kHD);
      #pragma unroll
      for (int p = 0; p < 8; ++p) {
        const int r  = p * 16 + (t >> 4);
        const int c8 = (t & 15) * 8;
        async_ld_b128(ldsbase + (unsigned)(KS_OFF + r * 136 + c8) * 2u,
                      ((unsigned)r * (unsigned)kKVD + (unsigned)c8) * 2u, gk);
      }
    }
    for (int p = 0; p < 8; ++p) {
      const int r  = p * 16 + (t >> 4);
      const int c8 = (t & 15) * 8;
      U128 raw = *(const U128*)&Vb[(size_t)(kb * 128 + r) * kKVD + kvh * kHD + c8];
      const unsigned short* e = (const unsigned short*)&raw;
      #pragma unroll
      for (int j = 0; j < 8; ++j) Vts[(c8 + j) * 136 + r] = e[j];  // transpose
    }
    wait_async0();    // own-wave async (Q on first iter, K always) complete
    __syncthreads();  // cross-wave visibility of Q/K/V tiles

    // ---- scores S = Q * K^T : per wave 16 rows x 128 keys ----
    v8f sc[8];
    #pragma unroll
    for (int nt = 0; nt < 8; ++nt) sc[nt] = zero8();
    for (int kk = 0; kk < 4; ++kk) {
      const v16bf aq = frag_a(Qs, 136, wid * 16, kk * 32, lane);
      #pragma unroll
      for (int nt = 0; nt < 8; ++nt) {
        const v16bf bk = frag_b(Ks, 136, nt * 16, kk * 32, lane);
        sc[nt] = wmma_bf16(aq, bk, sc[nt]);
      }
    }
    // Scale + causal mask on the diagonal block only.
    const int rowg = qb * 128 + wid * 16 + rh;
    #pragma unroll
    for (int nt = 0; nt < 8; ++nt) {
      const int colg = kb * 128 + nt * 16 + cn;
      #pragma unroll
      for (int r = 0; r < 8; ++r) {
        float sv = sc[nt][r] * scale;
        if (kb == qb && colg > rowg + r) sv = -3.0e38f;
        sc[nt][r] = sv;
      }
    }
    // ---- online softmax (rows 0-7 in lanes 0-15, rows 8-15 in lanes 16-31) --
    #pragma unroll
    for (int r = 0; r < 8; ++r) {
      float mx = sc[0][r];
      #pragma unroll
      for (int nt = 1; nt < 8; ++nt) mx = fmaxf(mx, sc[nt][r]);
      #pragma unroll
      for (int d = 1; d < 16; d <<= 1) mx = fmaxf(mx, __shfl_xor(mx, d, 32));
      const float nm    = fmaxf(mrow[r], mx);
      const float alpha = __expf(mrow[r] - nm);
      mrow[r] = nm;
      lrow[r] *= alpha;
      #pragma unroll
      for (int nt = 0; nt < 8; ++nt) accO[nt][r] *= alpha;
    }
    float psum[8];
    #pragma unroll
    for (int r = 0; r < 8; ++r) psum[r] = 0.f;
    #pragma unroll
    for (int nt = 0; nt < 8; ++nt)
      #pragma unroll
      for (int r = 0; r < 8; ++r) {
        const float pv = __expf(sc[nt][r] - mrow[r]);
        sc[nt][r] = pv;
        psum[r] += pv;
      }
    #pragma unroll
    for (int r = 0; r < 8; ++r) {
      float sum = psum[r];
      #pragma unroll
      for (int d = 1; d < 16; d <<= 1) sum += __shfl_xor(sum, d, 32);
      lrow[r] += sum;
    }
    // Re-layout P (C-layout regs -> A-fragment layout) through per-wave LDS.
    #pragma unroll
    for (int nt = 0; nt < 8; ++nt)
      #pragma unroll
      for (int r = 0; r < 8; ++r)
        Ps[(wid * 16 + rh + r) * 136 + nt * 16 + cn] = f32_to_bf16(sc[nt][r]);

    // ---- O += P * V ----
    for (int kk = 0; kk < 4; ++kk) {
      const v16bf ap = frag_a(Ps, 136, wid * 16, kk * 32, lane);
      #pragma unroll
      for (int nt = 0; nt < 8; ++nt) {
        const v16bf bv = frag_b(Vts, 136, nt * 16, kk * 32, lane);
        accO[nt] = wmma_bf16(ap, bv, accO[nt]);
      }
    }
  }

  // Epilogue: O /= l, store bf16 to attn buffer [s][head*128+hd].
  #pragma unroll
  for (int r = 0; r < 8; ++r) {
    const float rinv = 1.0f / lrow[r];
    const int row = qb * 128 + wid * 16 + rh + r;
    #pragma unroll
    for (int nt = 0; nt < 8; ++nt) {
      const int col = h * kHD + nt * 16 + cn;
      Ob[(size_t)row * kQD + col] = f32_to_bf16(accO[nt][r] * rinv);
    }
  }
}

// ---------------------------------------------------------------------------
extern "C" void kernel_launch(void* const* d_in, const int* in_sizes, int n_in,
                              void* d_out, int out_size, void* d_ws,
                              size_t ws_size, hipStream_t stream) {
  (void)in_sizes; (void)n_in; (void)out_size; (void)ws_size;
  const float* hidden = (const float*)d_in[0];
  const float* wq = (const float*)d_in[1];
  const float* wk = (const float*)d_in[2];
  const float* wv = (const float*)d_in[3];
  const float* wo = (const float*)d_in[4];
  float* out = (float*)d_out;

  // Workspace (bf16): q[2048x4096] k[2048x1024] v[2048x1024] attn[2048x4096]
  unsigned short* qbuf = (unsigned short*)d_ws;
  unsigned short* kbuf = qbuf + (size_t)kS * kQD;
  unsigned short* vbuf = kbuf + (size_t)kS * kKVD;
  unsigned short* abuf = vbuf + (size_t)kS * kKVD;

  const dim3 blk(256);
  const size_t gemm_smem = 2u * 10240u * sizeof(unsigned short);  // 40,960 B
  // QKV projections (f32 in -> bf16 out)
  gemm_tile128<false, false>
      <<<dim3(kQD / 128, kS / 128), blk, gemm_smem, stream>>>(hidden, wq, qbuf,
                                                              kS, kQD, kH);
  gemm_tile128<false, false>
      <<<dim3(kKVD / 128, kS / 128), blk, gemm_smem, stream>>>(hidden, wk, kbuf,
                                                               kS, kKVD, kH);
  gemm_tile128<false, false>
      <<<dim3(kKVD / 128, kS / 128), blk, gemm_smem, stream>>>(hidden, wv, vbuf,
                                                               kS, kKVD, kH);
  // RoPE on q and k
  rope_kernel<<<dim3((kS * kNQ * 64) / 256), blk, 0, stream>>>(qbuf, kNQ);
  rope_kernel<<<dim3((kS * kNKV * 64) / 256), blk, 0, stream>>>(kbuf, kNKV);
  // Attention (139,264 B dynamic LDS per workgroup — CDNA5 WGP has 320 KB)
  const size_t smem = 4u * 128u * 136u * sizeof(unsigned short);
  (void)hipFuncSetAttribute((const void*)attn_kernel,
                            hipFuncAttributeMaxDynamicSharedMemorySize,
                            (int)smem);
  attn_kernel<<<dim3(kS / kBlk, kNQ), blk, smem, stream>>>(qbuf, kbuf, vbuf,
                                                           abuf);
  // Output projection (bf16 A via async-LDS -> f32 out)
  gemm_tile128<true, true>
      <<<dim3(kH / 128, kS / 128), blk, gemm_smem, stream>>>(abuf, wo, out, kS,
                                                             kH, kQD);
}
